// DCNv4_75368086110976
// MI455X (gfx1250) — compile-verified, matrix operand order checked by reading, not run on hardware
//
#include <hip/hip_runtime.h>
#include <hip/hip_bf16.h>
#include <math.h>

// ---------------------------------------------------------------------------
// DCNv4 forward on MI455X (gfx1250, wave32).
//   K0: weights -> bf16, transposed Wt[col][k]  (one-time, L2-resident)
//   K1: val = xf @ w_value + b  -> (N,G,P,Cg)   [v_wmma_f32_16x16x32_bf16]
//   K1: om  = xf @ w_off  + b   -> (N,P,256)    [v_wmma_f32_16x16x32_bf16]
//   K2: deformable bilinear gather -> mid (N,P,C)  [fp32, 64B-coalesced]
//   K3: out = mid @ w_out + b, BN, SiLU -> NCHW [WMMA, fused epilogue]
// All addressing 32-bit; B fragments are 2x global_load_b128 from the
// pre-transposed bf16 weights; A fragments are 2x ds_load_b128 from LDS.
// ---------------------------------------------------------------------------

typedef __attribute__((ext_vector_type(16))) __bf16 v16bf;
typedef __attribute__((ext_vector_type(8)))  __bf16 v8bf;
typedef __attribute__((ext_vector_type(8)))  float  v8f;

#define G_   8
#define KK   9
#define OM_  32
#define C_   128
#define CG   16
#define H_   64
#define W_   64
#define P_   4096          // H*W
#define NB   4
#define NROWS (NB * P_)    // 16384
#define LDA  136           // 128 + 8 pad; 272B rows keep 16B alignment

// Gather one 16-element bf16 fragment: two contiguous 16B runs (K=+0..7 and
// K=+16..23 relative to p), matching the ISA 16-bit A/B VGPR layout.
template <typename PT>
__device__ inline v16bf frag16(const PT* p) {
    v8bf lo = *(const v8bf*)(p);
    v8bf hi = *(const v8bf*)(p + 16);
    v16bf f;
#pragma unroll
    for (int i = 0; i < 8; ++i) { f[i] = lo[i]; f[8 + i] = hi[i]; }
    return f;
}

// ---- K0: convert weight (K=128 rows x ncols, row-major fp32) into
// transposed bf16: wt[col*128 + k].
__global__ __launch_bounds__(256)
void prep_w(const float* __restrict__ w, __bf16* __restrict__ wt, int ncols) {
    const int tid = blockIdx.x * 256 + threadIdx.x;   // col*128 + k
    if (tid >= ncols * C_) return;
    const int k = tid & 127, col = tid >> 7;
    wt[tid] = (__bf16)w[k * ncols + col];
}

// ---- K1: GEMM, A = xf from NCHW x. Block = 4 waves; each wave does two
// 16x16 tiles (cols wave*32 and wave*32+16) sharing one A fragment.
// mode 0: write val as (N,G,P,Cg).  mode 1: write row-major (rows x n_out).
__global__ __launch_bounds__(128)
void gemm_xw(const float* __restrict__ x, const __bf16* __restrict__ wt,
             const float* __restrict__ bias, float* __restrict__ out,
             int n_out, int mode) {
    __shared__ __align__(16) __bf16 Alds[16 * LDA];
    const int r0 = blockIdx.x * 16;
    const int n  = r0 >> 12;            // P = 4096
    const int p0 = r0 & (P_ - 1);
    const int t  = threadIdx.x;
    {   // stage A: thread t owns channel c=t, 16 contiguous pixels
        const float* src = x + (n * C_ + t) * P_ + p0;
#pragma unroll
        for (int q = 0; q < 4; ++q) {
            float4 v = *(const float4*)(src + q * 4);
            Alds[(q * 4 + 0) * LDA + t] = (__bf16)v.x;
            Alds[(q * 4 + 1) * LDA + t] = (__bf16)v.y;
            Alds[(q * 4 + 2) * LDA + t] = (__bf16)v.z;
            Alds[(q * 4 + 3) * LDA + t] = (__bf16)v.w;
        }
    }
    __syncthreads();
    const int lane = t & 31, hi = lane >> 4, m = lane & 15;
    const int nc0 = blockIdx.y * 128 + (t >> 5) * 32 + m;   // tile0 column
    const int nc1 = nc0 + 16;                               // tile1 column
    const __bf16* b0p = wt + nc0 * C_ + 8 * hi;
    const __bf16* b1p = wt + nc1 * C_ + 8 * hi;
    const __bf16* ap  = Alds + m * LDA + 8 * hi;
    v8f acc0 = {}, acc1 = {};
#pragma unroll
    for (int kc = 0; kc < C_; kc += 32) {
        v16bf a  = frag16(ap + kc);
        v16bf b0 = frag16(b0p + kc);
        v16bf b1 = frag16(b1p + kc);
        acc0 = __builtin_amdgcn_wmma_f32_16x16x32_bf16(false, a, false, b0,
                                                       (short)0, acc0, false, false);
        acc1 = __builtin_amdgcn_wmma_f32_16x16x32_bf16(false, a, false, b1,
                                                       (short)0, acc1, false, false);
    }
#pragma unroll
    for (int tile = 0; tile < 2; ++tile) {
        const int ncol = tile ? nc1 : nc0;
        const v8f& acc = tile ? acc1 : acc0;
        const float bv = bias[ncol];
        if (mode == 0) {             // (N,G,P,Cg)
            const int g = ncol >> 4, cg = ncol & 15;
            float* dst = out + ((n * G_ + g) * P_ + p0) * CG + cg;
#pragma unroll
            for (int r = 0; r < 8; ++r) dst[(r + 8 * hi) * CG] = acc[r] + bv;
        } else {                     // row-major (rows, n_out)
            float* dst = out + r0 * n_out + ncol;
#pragma unroll
            for (int r = 0; r < 8; ++r) dst[(r + 8 * hi) * n_out] = acc[r] + bv;
        }
    }
}

// ---- K2: deformable bilinear sampling. One thread per (n,p,g,cg):
// every corner fetch is 64B contiguous across the 16 cg lanes.
__global__ __launch_bounds__(256)
void dcn_sample(const float* __restrict__ val, const float* __restrict__ om,
                float* __restrict__ mid) {
    const int tid = blockIdx.x * 256 + threadIdx.x;
    const int cg = tid & 15;
    const int g  = (tid >> 4) & 7;
    const int np = tid >> 7;           // n*P + p
    const int p  = np & (P_ - 1);
    const int n  = np >> 12;
    const int py = p >> 6, px = p & 63;
    const float* ob = om + np * (G_ * OM_) + g * OM_;
    const float* vb = val + ((n * G_ + g) * P_) * CG + cg;
    __builtin_prefetch(ob + G_ * OM_, 0, 1);      // next pixel's offset row
    float ov[28];                                 // 18 offsets + 9 masks
    {
        const float4* o4 = (const float4*)ob;
#pragma unroll
        for (int q = 0; q < 7; ++q) {
            float4 v = o4[q];
            ov[4 * q + 0] = v.x; ov[4 * q + 1] = v.y;
            ov[4 * q + 2] = v.z; ov[4 * q + 3] = v.w;
        }
    }
    float acc = 0.f;
#pragma unroll
    for (int k = 0; k < KK; ++k) {
        const float dx = ov[2 * k], dy = ov[2 * k + 1], mk = ov[2 * KK + k];
        const float sy = (float)(py + (k / 3) - 1) + dy;
        const float sx = (float)(px + (k % 3) - 1) + dx;
        const float y0f = floorf(sy), x0f = floorf(sx);
        const float ty = sy - y0f, tx = sx - x0f;
        const int y0 = (int)y0f, x0 = (int)x0f;
        float s = 0.f;
#pragma unroll
        for (int c = 0; c < 4; ++c) {
            const int iy = y0 + (c >> 1), ix = x0 + (c & 1);
            const float wy = (c >> 1) ? ty : 1.f - ty;
            const float wx = (c & 1) ? tx : 1.f - tx;
            const bool ok = (iy >= 0) & (iy < H_) & (ix >= 0) & (ix < W_);
            const int cy = iy < 0 ? 0 : (iy > H_ - 1 ? H_ - 1 : iy);
            const int cx = ix < 0 ? 0 : (ix > W_ - 1 ? W_ - 1 : ix);
            const float v = vb[(cy * W_ + cx) * CG];
            s += (ok ? wy * wx : 0.f) * v;
        }
        acc += mk * s;
    }
    mid[np * C_ + g * CG + cg] = acc;
}

// ---- K3: out-projection GEMM + BatchNorm + SiLU, NCHW store.
__global__ __launch_bounds__(128)
void gemm_out_bn(const float* __restrict__ mid, const __bf16* __restrict__ wt,
                 const float* __restrict__ bias, const float* __restrict__ gamma,
                 const float* __restrict__ beta, const float* __restrict__ mean,
                 const float* __restrict__ var, float* __restrict__ out) {
    __shared__ __align__(16) __bf16 Alds[16 * LDA];
    const int r0 = blockIdx.x * 16;
    const int n  = r0 >> 12;
    const int p0 = r0 & (P_ - 1);
    const int t  = threadIdx.x;
    {   // stage A panel from row-major mid
        const int row = t >> 3, seg = t & 7;
        const float* src = mid + (r0 + row) * C_ + seg * 16;
        __bf16* dl = Alds + row * LDA + seg * 16;
#pragma unroll
        for (int q = 0; q < 4; ++q) {
            float4 v = *(const float4*)(src + q * 4);
            dl[q * 4 + 0] = (__bf16)v.x; dl[q * 4 + 1] = (__bf16)v.y;
            dl[q * 4 + 2] = (__bf16)v.z; dl[q * 4 + 3] = (__bf16)v.w;
        }
    }
    __syncthreads();
    const int lane = t & 31, hi = lane >> 4, m = lane & 15;
    const int nc0 = (t >> 5) * 32 + m;       // output channel, tile0
    const int nc1 = nc0 + 16;
    const __bf16* b0p = wt + nc0 * C_ + 8 * hi;
    const __bf16* b1p = wt + nc1 * C_ + 8 * hi;
    const __bf16* ap  = Alds + m * LDA + 8 * hi;
    v8f acc0 = {}, acc1 = {};
#pragma unroll
    for (int kc = 0; kc < C_; kc += 32) {
        v16bf a  = frag16(ap + kc);
        v16bf b0 = frag16(b0p + kc);
        v16bf b1 = frag16(b1p + kc);
        acc0 = __builtin_amdgcn_wmma_f32_16x16x32_bf16(false, a, false, b0,
                                                       (short)0, acc0, false, false);
        acc1 = __builtin_amdgcn_wmma_f32_16x16x32_bf16(false, a, false, b1,
                                                       (short)0, acc1, false, false);
    }
#pragma unroll
    for (int tile = 0; tile < 2; ++tile) {
        const int c = tile ? nc1 : nc0;
        const v8f& acc = tile ? acc1 : acc0;
        const float inv = gamma[c] * rsqrtf(var[c] + 1e-5f);
        const float shf = beta[c] - mean[c] * inv;
        const float bv  = bias[c];
        float* dst = out + (n * C_ + c) * P_ + p0;
#pragma unroll
        for (int r = 0; r < 8; ++r) {
            float y = (acc[r] + bv) * inv + shf;
            dst[r + 8 * hi] = y * (1.f / (1.f + __expf(-y)));    // SiLU
        }
    }
}

extern "C" void kernel_launch(void* const* d_in, const int* in_sizes, int n_in,
                              void* d_out, int out_size, void* d_ws, size_t ws_size,
                              hipStream_t stream) {
    const float* x       = (const float*)d_in[0];
    const float* w_value = (const float*)d_in[1];
    const float* b_value = (const float*)d_in[2];
    const float* w_off   = (const float*)d_in[3];
    const float* b_off   = (const float*)d_in[4];
    const float* w_out   = (const float*)d_in[5];
    const float* b_out   = (const float*)d_in[6];
    const float* gamma   = (const float*)d_in[7];
    const float* beta    = (const float*)d_in[8];
    const float* mean    = (const float*)d_in[9];
    const float* var     = (const float*)d_in[10];
    float* out = (float*)d_out;

    float*  val_ws = (float*)d_ws;                          //  8 MB
    float*  om_ws  = val_ws + NB * G_ * P_ * CG;            // 16 MB
    float*  mid_ws = om_ws + NB * P_ * (G_ * OM_);          //  8 MB
    __bf16* wvt    = (__bf16*)(mid_ws + NB * P_ * C_);      // 32 KB
    __bf16* wot    = wvt + C_ * C_;                         // 64 KB
    __bf16* wut    = wot + C_ * (G_ * OM_);                 // 32 KB

    prep_w<<<dim3((C_ * C_) / 256), dim3(256), 0, stream>>>(w_value, wvt, C_);
    prep_w<<<dim3((C_ * G_ * OM_) / 256), dim3(256), 0, stream>>>(w_off, wot, G_ * OM_);
    prep_w<<<dim3((C_ * C_) / 256), dim3(256), 0, stream>>>(w_out, wut, C_);

    gemm_xw<<<dim3(NROWS / 16, 1), dim3(128), 0, stream>>>(
        x, wvt, b_value, val_ws, /*n_out=*/C_, /*mode=*/0);
    gemm_xw<<<dim3(NROWS / 16, 2), dim3(128), 0, stream>>>(
        x, wot, b_off, om_ws, /*n_out=*/G_ * OM_, /*mode=*/1);
    dcn_sample<<<dim3((NB * P_ * G_ * CG) / 256), dim3(256), 0, stream>>>(
        val_ws, om_ws, mid_ws);
    gemm_out_bn<<<dim3(NROWS / 16, 1), dim3(128), 0, stream>>>(
        mid_ws, wut, b_out, gamma, beta, mean, var, out);
}
